// WindowAttention_42322607735266
// MI455X (gfx1250) — compile-verified, matrix operand order checked by reading, not run on hardware
//
#include <hip/hip_runtime.h>
#include <hip/hip_bf16.h>

typedef __attribute__((ext_vector_type(16))) _Float16 v16h;
typedef __attribute__((ext_vector_type(8)))  _Float16 v8h;
typedef __attribute__((ext_vector_type(8)))  float    v8f;

#define NTOK 49
#define CDIM 128
#define NHEADS 4

union F16Frag { v16h v; v8h h[2]; };

// Load a 16x32 f16 WMMA A-style fragment from a row-major matrix.
// Layout (ISA 7.12.2): lane(0-15)=row, halves0-7 = K kbase+sub..+7, halves8-15 = K kbase+sub+16..+23,
// sub = 8 for lanes 16-31.
static __device__ __forceinline__ v16h ld_frag(const _Float16* base, int row, int stride,
                                               int kbase, int lane) {
  const int sub = (lane >> 4) << 3;
  const _Float16* p = base + row * stride + kbase + sub;
  F16Frag f;
  f.h[0] = *(const v8h*)(p);
  f.h[1] = *(const v8h*)(p + 16);
  return f.v;
}

static __device__ __forceinline__ v8f zero_v8f() {
  v8f z;
#pragma unroll
  for (int i = 0; i < 8; ++i) z[i] = 0.0f;
  return z;
}

static __device__ __forceinline__ v8f wmma16(v16h a, v16h b, v8f c) {
  return __builtin_amdgcn_wmma_f32_16x16x32_f16(false, a, false, b, (short)0, c, false, false);
}

__global__ void convert_weights_kernel(const float* __restrict__ qkv_w,
                                       const float* __restrict__ proj_w,
                                       _Float16* __restrict__ wh) {
  int i = blockIdx.x * 256 + threadIdx.x;
  if (i < 384 * 128) wh[i] = (_Float16)qkv_w[i];
  else if (i < 384 * 128 + 128 * 128) wh[i] = (_Float16)proj_w[i - 384 * 128];
}

__global__ __launch_bounds__(256) void swin_window_attn(
    const float* __restrict__ x, const float* __restrict__ mask,
    const float* __restrict__ qkv_b, const float* __restrict__ proj_b,
    const float* __restrict__ bias_table,
    const _Float16* __restrict__ qkv_wh, const _Float16* __restrict__ proj_wh,
    float* __restrict__ out) {
  extern __shared__ _Float16 smem[];
  _Float16* sQK = smem;             // [2][4][64][32] q,k row-major per head (16384 halves)
  _Float16* sVt = smem + 16384;     // [4][32][64]    v transposed            (8192 halves)
  _Float16* sAh = smem + 24576;     // [4][64][64]    attn probabilities      (16384 halves)
  _Float16* sX  = smem + 40960;     // [64][128]      x (later reused as attn@v output)

  const int b    = blockIdx.x;
  const int tid  = threadIdx.x;
  const int lane = tid & 31;
  const int wv   = tid >> 5;        // wave id 0..7
  const int cl   = lane & 15;       // column-in-tile lane
  const int rh   = lane >> 4;       // row-half select
  const int w    = b & 63;          // mask window index

  // ---- stage 1: x -> LDS f16, zero-padded to 64 rows ----
  {
    const float* xb = x + (size_t)b * (NTOK * CDIM);
    for (int p = tid; p < 64 * 64; p += 256) {
      int row = p >> 6;
      int c2  = (p & 63) << 1;
      float f0 = 0.f, f1 = 0.f;
      if (row < NTOK) { f0 = xb[row * CDIM + c2]; f1 = xb[row * CDIM + c2 + 1]; }
      sX[row * CDIM + c2]     = (_Float16)f0;
      sX[row * CDIM + c2 + 1] = (_Float16)f1;
    }
  }
  __syncthreads();

  // ---- stage 2: qkv = x @ qkv_w^T + qkv_b  (M=64, N=384, K=128) ----
  for (int nt = wv; nt < 24; nt += 8) {
    const int colg = nt * 16 + cl;           // output feature 0..383
    const float bq = qkv_b[colg];
    v8f acc[4];
#pragma unroll
    for (int mt = 0; mt < 4; ++mt) acc[mt] = zero_v8f();
#pragma unroll
    for (int kt = 0; kt < 4; ++kt) {
      v16h bfr = ld_frag(qkv_wh, colg, CDIM, kt * 32, lane);  // B-frag = rows of qkv_w
#pragma unroll
      for (int mt = 0; mt < 4; ++mt) {
        v16h afr = ld_frag(sX, mt * 16 + cl, CDIM, kt * 32, lane);
        acc[mt] = wmma16(afr, bfr, acc[mt]);
      }
    }
    const int j  = colg >> 7;                // 0=q 1=k 2=v
    const int hh = (colg >> 5) & 3;
    const int d  = colg & 31;
#pragma unroll
    for (int mt = 0; mt < 4; ++mt) {
      if (j == 2) {                          // store v transposed, packed b128
        v8h pv;
#pragma unroll
        for (int r = 0; r < 8; ++r) pv[r] = (_Float16)(acc[mt][r] + bq);
        *(v8h*)(sVt + (hh * 32 + d) * 64 + mt * 16 + rh * 8) = pv;
      } else {                               // q/k row-major per head
#pragma unroll
        for (int r = 0; r < 8; ++r) {
          int m = mt * 16 + rh * 8 + r;
          sQK[((j * 4 + hh) * 64 + m) * 32 + d] = (_Float16)(acc[mt][r] + bq);
        }
      }
    }
  }
  __syncthreads();

  // ---- stage 3: logits = scale*q@k^T + bias + mask, softmax -> sAh ----
  const float scale = 0.17677669529663687f;  // 1/sqrt(32)
  for (int p = wv; p < 16; p += 8) {
    const int h  = p >> 2;
    const int it = p & 3;
    v16h aq = ld_frag(sQK + h * 64 * 32, it * 16 + cl, 32, 0, lane);
    v8f c[4];
    int pn[4], ni[4];
#pragma unroll
    for (int jj = 0; jj < 4; ++jj) {
      v16h bk = ld_frag(sQK + (4 + h) * 64 * 32, jj * 16 + cl, 32, 0, lane);
      c[jj] = wmma16(aq, bk, zero_v8f());
      int n  = jj * 16 + cl;
      int nc = n < NTOK ? n : NTOK - 1;
      ni[jj] = nc;
      pn[jj] = (nc / 7) * 13 + (nc % 7);
    }
    const float* maskw = mask + (size_t)w * (NTOK * NTOK);
#pragma unroll
    for (int r = 0; r < 8; ++r) {
      int m  = it * 16 + rh * 8 + r;
      int mc = m < NTOK ? m : NTOK - 1;
      int pm = (mc / 7) * 13 + (mc % 7) + 84;   // (mh-nh+6)*13 + (mw-nw+6) = pm - pn
      float l[4];
#pragma unroll
      for (int jj = 0; jj < 4; ++jj) {
        int n = jj * 16 + cl;
        float v = c[jj][r] * scale
                + bias_table[(pm - pn[jj]) * NHEADS + h]
                + maskw[mc * NTOK + ni[jj]];
        l[jj] = (n < NTOK) ? v : -1e30f;
      }
      float mx = fmaxf(fmaxf(l[0], l[1]), fmaxf(l[2], l[3]));
#pragma unroll
      for (int off = 1; off < 16; off <<= 1) mx = fmaxf(mx, __shfl_xor(mx, off, 32));
      float s = 0.f;
#pragma unroll
      for (int jj = 0; jj < 4; ++jj) { l[jj] = __expf(l[jj] - mx); s += l[jj]; }
#pragma unroll
      for (int off = 1; off < 16; off <<= 1) s += __shfl_xor(s, off, 32);
      float inv = 1.0f / s;
#pragma unroll
      for (int jj = 0; jj < 4; ++jj) {
        int n = jj * 16 + cl;
        sAh[(h * 64 + m) * 64 + n] = (_Float16)(l[jj] * inv);
      }
    }
  }
  __syncthreads();

  // ---- stage 4: out_h = attn @ v  (per head M=64, N=32, K=64) -> sX [64][128] ----
  {
    const int it = wv >> 1;
    const int dt = wv & 1;
#pragma unroll
    for (int h = 0; h < 4; ++h) {
      v8f c = zero_v8f();
#pragma unroll
      for (int kk = 0; kk < 2; ++kk) {
        v16h a  = ld_frag(sAh + h * 64 * 64, it * 16 + cl, 64, kk * 32, lane);
        v16h bv = ld_frag(sVt + h * 32 * 64, dt * 16 + cl, 64, kk * 32, lane); // rows of v^T
        c = wmma16(a, bv, c);
      }
#pragma unroll
      for (int r = 0; r < 8; ++r) {
        int m = it * 16 + rh * 8 + r;
        sX[m * CDIM + h * 32 + dt * 16 + cl] = (_Float16)c[r];
      }
    }
  }
  __syncthreads();

  // ---- stage 5: out = o @ proj_w^T + proj_b  (M=64, N=128, K=128) ----
  {
    const int nt   = wv;
    const int colg = nt * 16 + cl;
    const float bp = proj_b[colg];
    v8f acc[4];
#pragma unroll
    for (int it = 0; it < 4; ++it) acc[it] = zero_v8f();
#pragma unroll
    for (int kt = 0; kt < 4; ++kt) {
      v16h bfr = ld_frag(proj_wh, colg, CDIM, kt * 32, lane);
#pragma unroll
      for (int it = 0; it < 4; ++it) {
        v16h afr = ld_frag(sX, it * 16 + cl, CDIM, kt * 32, lane);
        acc[it] = wmma16(afr, bfr, acc[it]);
      }
    }
    float* ob = out + (size_t)b * (NTOK * CDIM);
#pragma unroll
    for (int it = 0; it < 4; ++it) {
#pragma unroll
      for (int r = 0; r < 8; ++r) {
        int m = it * 16 + rh * 8 + r;
        if (m < NTOK) ob[m * CDIM + colg] = acc[it][r] + bp;
      }
    }
  }
}

extern "C" void kernel_launch(void* const* d_in, const int* in_sizes, int n_in,
                              void* d_out, int out_size, void* d_ws, size_t ws_size,
                              hipStream_t stream) {
  const float* x          = (const float*)d_in[0];
  const float* mask       = (const float*)d_in[1];
  const float* qkv_w      = (const float*)d_in[2];
  const float* qkv_b      = (const float*)d_in[3];
  const float* proj_w     = (const float*)d_in[4];
  const float* proj_b     = (const float*)d_in[5];
  const float* bias_table = (const float*)d_in[6];
  _Float16* wh = (_Float16*)d_ws;      // 49152 + 16384 halves = 128 KB scratch

  convert_weights_kernel<<<256, 256, 0, stream>>>(qkv_w, proj_w, wh);

  const size_t shmem = 49152 * sizeof(_Float16);   // 96 KB LDS per workgroup
  swin_window_attn<<<8192, 256, shmem, stream>>>(x, mask, qkv_b, proj_b, bias_table,
                                                 wh, wh + 384 * 128, (float*)d_out);
}
